// MemTransformerLM_16071767622335
// MI455X (gfx1250) — matrix operand order repaired and checked by reading, not compiled, not run on hardware
//
#include <hip/hip_runtime.h>
#include <math.h>

#define NLAYER 12
#define NHEAD  8
#define DH     64
#define DMODEL 512
#define DIFF   2048
#define QL     512
#define ML     512
#define KL     1024
#define BS     16
#define QKV3   1536

typedef _Float16 f16;
typedef f16   v8h  __attribute__((ext_vector_type(8)));
typedef f16   v16h __attribute__((ext_vector_type(16)));
typedef float v8f  __attribute__((ext_vector_type(8)));

__device__ __forceinline__ v8f wm(v16h a, v16h b, v8f c) {
  // D = A(16x32,f16)*B(32x16,f16) + C(16x16,f32)
  return __builtin_amdgcn_wmma_f32_16x16x32_f16(false, a, false, b, (short)0, c,
                                                false, false);
}
__device__ __forceinline__ void split(float x, f16& hi, f16& lo) {
  hi = (f16)x;
  lo = (f16)(x - (float)hi);
}

// ---------------- generic f32 -> hi/lo f16 plane converter ----------------
__global__ void k_split(const float* __restrict__ src, f16* __restrict__ dH,
                        f16* __restrict__ dL, size_t n) {
  size_t i = (size_t)blockIdx.x * 256 + threadIdx.x;
  if (i >= n) return;
  f16 h, l; split(src[i], h, l);
  dH[i] = h; dL[i] = l;
}

// ---------------- embedding gather -> planes (QL*BS, 768) ----------------
__global__ void k_embed(const int* __restrict__ pitch, const int* __restrict__ dur,
                        const int* __restrict__ chord,
                        const float* __restrict__ pew, const float* __restrict__ dew,
                        f16* __restrict__ eH, f16* __restrict__ eL) {
  int idx = blockIdx.x * 256 + threadIdx.x;
  if (idx >= QL * BS * 768) return;
  int t = idx / 768, col = idx % 768;
  float v;
  if (col < 128)       v = pew[pitch[t] * 128 + col];
  else if (col < 256)  v = dew[dur[t]   * 128 + (col - 128)];
  else { int c = (col - 256) >> 7, cc = (col - 256) & 127;
         v = pew[chord[t * 4 + c] * 128 + cc]; }
  f16 h, l; split(v, h, l);
  eH[idx] = h; eL[idx] = l;
}

// ---------------- positional encodings r (KL, DMODEL) -> planes ----------------
__global__ void k_pos(f16* __restrict__ rH, f16* __restrict__ rL) {
  int idx = blockIdx.x * 256 + threadIdx.x;
  if (idx >= KL * 256) return;
  int p = idx >> 8, f = idx & 255;
  float pos  = (float)(KL - 1 - p);
  float invf = powf(10000.0f, -((float)(2 * f)) / (float)DMODEL);
  float a = pos * invf;
  f16 h, l;
  int i1 = p * DMODEL + f, i2 = i1 + 256;
  split(sinf(a), h, l); rH[i1] = h; rL[i1] = l;
  split(cosf(a), h, l); rH[i2] = h; rL[i2] = l;
}

// ---------------- cat = [mems_l ; h] -> planes (KL*BS, DMODEL) ----------------
__global__ void k_cat(const float* __restrict__ mems_l, const float* __restrict__ hsrc,
                      f16* __restrict__ cH, f16* __restrict__ cL) {
  int idx = blockIdx.x * 256 + threadIdx.x;
  if (idx >= KL * BS * DMODEL) return;
  int row = idx / DMODEL;
  int j = row / BS;
  float v = (j < ML) ? mems_l[idx] : hsrc[idx - ML * BS * DMODEL];
  f16 h, l; split(v, h, l);
  cH[idx] = h; cL[idx] = l;
}

// ------------- split-f16 WMMA GEMM: C = A(MxK) @ B(NxK)^T (+bias) -------------
// A,B given as hi/lo f16 planes; 3-product scheme ~= f32 accuracy.
// OUTPLANES: write hi/lo f16 planes (oH/oL); else write f32 C.
template <bool RELU, bool OUTPLANES>
__global__ __launch_bounds__(32)
void k_gemm16(const f16* __restrict__ aH, const f16* __restrict__ aL,
              const f16* __restrict__ bH, const f16* __restrict__ bL,
              const float* __restrict__ bias,
              float* __restrict__ C, f16* __restrict__ oH, f16* __restrict__ oL,
              int M, int N, int K) {
  int lane = threadIdx.x;
  int m0 = blockIdx.y * 32, n0 = blockIdx.x * 32;
  int lrow = lane & 15, lh = lane >> 4;
  size_t ar0 = (size_t)(m0 + lrow) * K, ar1 = (size_t)(m0 + 16 + lrow) * K;
  size_t br0 = (size_t)(n0 + lrow) * K, br1 = (size_t)(n0 + 16 + lrow) * K;
  v8f c00 = {}, c01 = {}, c10 = {}, c11 = {};
  for (int k0 = 0; k0 < K; k0 += 32) {
    int ka = k0 + 16 * lh, kb = k0 + 16 * lh + 8, kn = k0 + 16 * lh;
    v16h a0h = __builtin_shufflevector(*(const v8h*)(aH + ar0 + ka), *(const v8h*)(aH + ar0 + kb),
                                       0,1,2,3,4,5,6,7,8,9,10,11,12,13,14,15);
    v16h a0l = __builtin_shufflevector(*(const v8h*)(aL + ar0 + ka), *(const v8h*)(aL + ar0 + kb),
                                       0,1,2,3,4,5,6,7,8,9,10,11,12,13,14,15);
    v16h a1h = __builtin_shufflevector(*(const v8h*)(aH + ar1 + ka), *(const v8h*)(aH + ar1 + kb),
                                       0,1,2,3,4,5,6,7,8,9,10,11,12,13,14,15);
    v16h a1l = __builtin_shufflevector(*(const v8h*)(aL + ar1 + ka), *(const v8h*)(aL + ar1 + kb),
                                       0,1,2,3,4,5,6,7,8,9,10,11,12,13,14,15);
    v16h b0h = *(const v16h*)(bH + br0 + kn);
    v16h b0l = *(const v16h*)(bL + br0 + kn);
    v16h b1h = *(const v16h*)(bH + br1 + kn);
    v16h b1l = *(const v16h*)(bL + br1 + kn);
    c00 = wm(a0h, b0h, c00); c00 = wm(a0h, b0l, c00); c00 = wm(a0l, b0h, c00);
    c01 = wm(a0h, b1h, c01); c01 = wm(a0h, b1l, c01); c01 = wm(a0l, b1h, c01);
    c10 = wm(a1h, b0h, c10); c10 = wm(a1h, b0l, c10); c10 = wm(a1l, b0h, c10);
    c11 = wm(a1h, b1h, c11); c11 = wm(a1h, b1l, c11); c11 = wm(a1l, b1h, c11);
  }
  int ne0 = n0 + lrow, ne1 = ne0 + 16;
  float bi0 = bias ? bias[ne0] : 0.f;
  float bi1 = bias ? bias[ne1] : 0.f;
  for (int rr = 0; rr < 8; ++rr) {
    int mra = m0 + rr + 8 * lh, mrb = mra + 16;
    float v0 = c00[rr] + bi0, v1 = c01[rr] + bi1, v2 = c10[rr] + bi0, v3 = c11[rr] + bi1;
    if (RELU) { v0 = fmaxf(v0, 0.f); v1 = fmaxf(v1, 0.f); v2 = fmaxf(v2, 0.f); v3 = fmaxf(v3, 0.f); }
    if (OUTPLANES) {
      f16 h, l;
      split(v0, h, l); oH[(size_t)mra * N + ne0] = h; oL[(size_t)mra * N + ne0] = l;
      split(v1, h, l); oH[(size_t)mra * N + ne1] = h; oL[(size_t)mra * N + ne1] = l;
      split(v2, h, l); oH[(size_t)mrb * N + ne0] = h; oL[(size_t)mrb * N + ne0] = l;
      split(v3, h, l); oH[(size_t)mrb * N + ne1] = h; oL[(size_t)mrb * N + ne1] = l;
    } else {
      C[(size_t)mra * N + ne0] = v0;
      C[(size_t)mra * N + ne1] = v1;
      C[(size_t)mrb * N + ne0] = v2;
      C[(size_t)mrb * N + ne1] = v3;
    }
  }
}

// ---------------- h = LayerNorm(h + res); also emit hi/lo planes ----------------
__global__ __launch_bounds__(32)
void k_addln(float* __restrict__ h, const float* __restrict__ res,
             const float* __restrict__ g, const float* __restrict__ b,
             f16* __restrict__ hH, f16* __restrict__ hL) {
  int row = blockIdx.x, lane = threadIdx.x;
  float x[16];
  float s = 0.f;
  const float* hr = h + (size_t)row * DMODEL;
  const float* rr = res + (size_t)row * DMODEL;
  for (int i = 0; i < 16; ++i) {
    int c = lane + 32 * i;
    x[i] = hr[c] + rr[c];
    s += x[i];
  }
  for (int o = 16; o >= 1; o >>= 1) s += __shfl_xor(s, o, 32);
  float mu = s * (1.f / DMODEL);
  float vs = 0.f;
  for (int i = 0; i < 16; ++i) { float d = x[i] - mu; vs += d * d; }
  for (int o = 16; o >= 1; o >>= 1) vs += __shfl_xor(vs, o, 32);
  float rstd = 1.f / sqrtf(vs * (1.f / DMODEL) + 1e-5f);
  float* out = h + (size_t)row * DMODEL;
  for (int i = 0; i < 16; ++i) {
    int c = lane + 32 * i;
    float y = (x[i] - mu) * rstd * g[c] + b[c];
    out[c] = y;
    f16 yh, yl; split(y, yh, yl);
    hH[(size_t)row * DMODEL + c] = yh;
    hL[(size_t)row * DMODEL + c] = yl;
  }
}

// ---------------- fused rel-attention, split-f16 WMMA ----------------
// score[i,j] = scale*((q+rwb)_i.k_j + (q+rrb)_i.rk_{j-i+511}); mask j > i+ML
__global__ __launch_bounds__(32)
void k_attn(const f16* __restrict__ whH, const f16* __restrict__ whL, // (KL*BS,1536)
            const f16* __restrict__ rkH, const f16* __restrict__ rkL, // (KL, DMODEL)
            const float* __restrict__ rwb, const float* __restrict__ rrb,
            f16* __restrict__ avH, f16* __restrict__ avL) {           // (QL*BS, DMODEL)
  __shared__ float S[16 * 1024];
  const int lane = threadIdx.x;
  const int i0 = blockIdx.x * 16;
  const int b  = blockIdx.y;
  const int n  = blockIdx.z;
  const int lrow = lane & 15, lh = lane >> 4;
  const float scale = 0.125f;

  // A-fragments of (q + r_w_bias) and (q + r_r_bias): 2 chunks of K=32
  v16h awh[2], awl[2], arh[2], arl[2];
  {
    size_t qoff = ((size_t)(ML + i0 + lrow) * BS + b) * QKV3 + n * DH;
    const float* wbp = rwb + n * DH;
    const float* rbp = rrb + n * DH;
    for (int c = 0; c < 2; ++c)
      for (int gg = 0; gg < 2; ++gg) {
        int ks = 32 * c + 8 * gg + 16 * lh;
        v8h qh = *(const v8h*)(whH + qoff + ks);
        v8h ql = *(const v8h*)(whL + qoff + ks);
        for (int i = 0; i < 8; ++i) {
          float q = (float)qh[i] + (float)ql[i];
          f16 h, l;
          split(q + wbp[ks + i], h, l); awh[c][8 * gg + i] = h; awl[c][8 * gg + i] = l;
          split(q + rbp[ks + i], h, l); arh[c][8 * gg + i] = h; arl[c][8 * gg + i] = l;
        }
      }
  }

  // -------- score pass: 64 j-tiles --------
  for (int jt = 0; jt < 64; ++jt) {
    int j0 = jt * 16;
    v8f ac = {}, br0acc = {}, br1acc = {};
    size_t krow = ((size_t)(j0 + lrow) * BS + b) * QKV3 + DMODEL + n * DH;
    int jp0 = j0 - i0 + 496;            // multiple of 16, >= 0
    int r0row = jp0 + lrow, r1row = r0row + 16;
    bool ok0 = (r0row < KL), ok1 = (r1row < KL);
    size_t r0off = (size_t)r0row * DMODEL + n * DH;
    size_t r1off = (size_t)r1row * DMODEL + n * DH;
    for (int c = 0; c < 2; ++c) {
      int kn = 32 * c + 16 * lh;
      v16h kh = *(const v16h*)(whH + krow + kn);
      v16h kl = *(const v16h*)(whL + krow + kn);
      ac = wm(awh[c], kh, ac); ac = wm(awh[c], kl, ac); ac = wm(awl[c], kh, ac);
      v16h r0h = {}, r0l = {}, r1h = {}, r1l = {};
      if (ok0) { r0h = *(const v16h*)(rkH + r0off + kn); r0l = *(const v16h*)(rkL + r0off + kn); }
      if (ok1) { r1h = *(const v16h*)(rkH + r1off + kn); r1l = *(const v16h*)(rkL + r1off + kn); }
      br0acc = wm(arh[c], r0h, br0acc); br0acc = wm(arh[c], r0l, br0acc); br0acc = wm(arl[c], r0h, br0acc);
      br1acc = wm(arh[c], r1h, br1acc); br1acc = wm(arh[c], r1l, br1acc); br1acc = wm(arl[c], r1h, br1acc);
    }
    // blend shifted BD diagonal band, mask, store to LDS
    for (int rr = 0; rr < 8; ++rr) {
      int mr = rr + 8 * lh;
      int jp_rel = lrow - mr + 15;       // 0..30
      int src = (jp_rel & 15) | (lh << 4);
      float v0 = __shfl(br0acc[rr], src, 32);
      float v1 = __shfl(br1acc[rr], src, 32);
      float bd = (jp_rel >= 16) ? v1 : v0;
      float sc = (ac[rr] + bd) * scale;
      int j_abs = j0 + lrow;
      int i_abs = i0 + mr;
      if (j_abs > i_abs + ML) sc = -1e30f;
      S[mr * 1024 + j_abs] = sc;
    }
  }
  __syncthreads();

  // -------- softmax (unnormalized exp; defer /sum past PV) --------
  float* Srow = S + lrow * 1024 + lh * 512;
  float mx = -3.0e38f;
  for (int j = 0; j < 512; ++j) mx = fmaxf(mx, Srow[j]);
  mx = fmaxf(mx, __shfl_xor(mx, 16, 32));
  float sum = 0.f;
  for (int j = 0; j < 512; ++j) {
    float e = expf(Srow[j] - mx);
    Srow[j] = e;
    sum += e;
  }
  sum += __shfl_xor(sum, 16, 32);        // lane L: rowsum of row L%16
  __syncthreads();

  // -------- PV: acc (16 x 64) in 4 WMMA tiles, K=1024 in 32 chunks --------
  v8f pv0 = {}, pv1 = {}, pv2 = {}, pv3 = {};
  for (int kc = 0; kc < 32; ++kc) {
    // A (probs) from LDS f32, split on the fly
    v16h aph, apl;
    for (int gg = 0; gg < 2; ++gg) {
      const float* sp = S + lrow * 1024 + 32 * kc + 8 * gg + 16 * lh;
      float4 x0 = *(const float4*)(sp);
      float4 x1 = *(const float4*)(sp + 4);
      float xs[8] = {x0.x, x0.y, x0.z, x0.w, x1.x, x1.y, x1.z, x1.w};
      for (int i = 0; i < 8; ++i) {
        f16 h, l; split(xs[i], h, l);
        aph[8 * gg + i] = h; apl[8 * gg + i] = l;
      }
    }
    // B (V) gathers: rows k = 32*kc + 16*lh + r2, col = 2*DMODEL + n*64 + dt*16 + lrow
    int kr0 = 32 * kc + 16 * lh;
    size_t vbase = ((size_t)kr0 * BS + b) * QKV3 + 2 * DMODEL + n * DH + lrow;
    v16h bh0, bl0, bh1, bl1, bh2, bl2, bh3, bl3;
    for (int r2 = 0; r2 < 16; ++r2) {
      size_t off = vbase + (size_t)r2 * BS * QKV3;
      bh0[r2] = whH[off];      bl0[r2] = whL[off];
      bh1[r2] = whH[off + 16]; bl1[r2] = whL[off + 16];
      bh2[r2] = whH[off + 32]; bl2[r2] = whL[off + 32];
      bh3[r2] = whH[off + 48]; bl3[r2] = whL[off + 48];
    }
    pv0 = wm(aph, bh0, pv0); pv0 = wm(aph, bl0, pv0); pv0 = wm(apl, bh0, pv0);
    pv1 = wm(aph, bh1, pv1); pv1 = wm(aph, bl1, pv1); pv1 = wm(apl, bh1, pv1);
    pv2 = wm(aph, bh2, pv2); pv2 = wm(aph, bl2, pv2); pv2 = wm(apl, bh2, pv2);
    pv3 = wm(aph, bh3, pv3); pv3 = wm(aph, bl3, pv3); pv3 = wm(apl, bh3, pv3);
  }
  for (int rr = 0; rr < 8; ++rr) {
    int mr = rr + 8 * lh;
    float inv = 1.f / __shfl(sum, mr, 32);
    size_t o = ((size_t)(i0 + mr) * BS + b) * DMODEL + n * DH + lrow;
    f16 h, l;
    split(pv0[rr] * inv, h, l); avH[o]      = h; avL[o]      = l;
    split(pv1[rr] * inv, h, l); avH[o + 16] = h; avL[o + 16] = l;
    split(pv2[rr] * inv, h, l); avH[o + 32] = h; avL[o + 32] = l;
    split(pv3[rr] * inv, h, l); avH[o + 48] = h; avL[o + 48] = l;
  }
}

// ======================================================================
extern "C" void kernel_launch(void* const* d_in, const int* in_sizes, int n_in,
                              void* d_out, int out_size, void* d_ws, size_t ws_size,
                              hipStream_t stream) {
  const int*   pitch = (const int*)d_in[0];
  const int*   dur   = (const int*)d_in[1];
  const int*   chord = (const int*)d_in[2];
  const float* mems  = (const float*)d_in[3];
  const float* pew   = (const float*)d_in[4];
  const float* dew   = (const float*)d_in[5];
  const float* projw = (const float*)d_in[6];
  const float* projb = (const float*)d_in[7];
  const float* rwb   = (const float*)d_in[8];
  const float* rrb   = (const float*)d_in[9];
  const float* qkvw  = (const float*)d_in[10];
  const float* rw    = (const float*)d_in[11];
  const float* ow    = (const float*)d_in[12];
  const float* ln1g  = (const float*)d_in[13];
  const float* ln1b  = (const float*)d_in[14];
  const float* ff1w  = (const float*)d_in[15];
  const float* ff1b  = (const float*)d_in[16];
  const float* ff2w  = (const float*)d_in[17];
  const float* ff2b  = (const float*)d_in[18];
  const float* ln2g  = (const float*)d_in[19];
  const float* ln2b  = (const float*)d_in[20];

  float* h = (float*)d_out;             // (QL*BS, DMODEL) persists across layers

  // ---- workspace layout: each "plane buffer" of N elems = N f16 hi + N f16 lo
  const size_t NPROJW = (size_t)DMODEL * 768;
  const size_t NQKVW  = (size_t)NLAYER * QKV3 * DMODEL;
  const size_t NRWW   = (size_t)NLAYER * DMODEL * DMODEL;
  const size_t NOWW   = NRWW;
  const size_t NFF1   = (size_t)NLAYER * DIFF * DMODEL;
  const size_t NFF2   = NFF1;
  const size_t NR     = (size_t)KL * DMODEL;
  const size_t NCAT   = (size_t)KL * BS * DMODEL;
  const size_t NWH    = (size_t)KL * BS * QKV3;
  const size_t NRK    = NR;
  const size_t NAV    = (size_t)QL * BS * DMODEL;
  const size_t NH     = NAV;
  const size_t NFFM   = (size_t)QL * BS * DIFF;
  const size_t NEMB   = (size_t)QL * BS * 768;

  float* cur = (float*)d_ws;
  f16 *projwH, *projwL, *qkvwH, *qkvwL, *rwwH, *rwwL, *owwH, *owwL;
  f16 *ff1wH, *ff1wL, *ff2wH, *ff2wL;
  f16 *rH, *rL, *catH, *catL, *whH, *whL, *rkH, *rkL, *avH, *avL, *hbH, *hbL, *ffmH, *ffmL;
  auto allocP = [&](size_t nels, f16*& H, f16*& Lo) {
    H = (f16*)cur; Lo = H + nels; cur += nels;   // nels floats == 2*nels f16
  };
  allocP(NPROJW, projwH, projwL);
  allocP(NQKVW,  qkvwH,  qkvwL);
  allocP(NRWW,   rwwH,   rwwL);
  allocP(NOWW,   owwH,   owwL);
  allocP(NFF1,   ff1wH,  ff1wL);
  allocP(NFF2,   ff2wH,  ff2wL);
  allocP(NR,     rH,     rL);
  allocP(NCAT,   catH,   catL);
  allocP(NWH,    whH,    whL);
  allocP(NRK,    rkH,    rkL);
  allocP(NAV,    avH,    avL);
  allocP(NH,     hbH,    hbL);
  allocP(NFFM,   ffmH,   ffmL);
  float* tmp = cur;                      // QL*BS*DMODEL f32
  // emb planes alias the (not-yet-used) ffmid region
  f16* embH = ffmH;
  f16* embL = embH + NEMB;

  // ---- one-time (per launch) weight splits ----
  auto splitLaunch = [&](const float* s, f16* dH, f16* dL, size_t n) {
    k_split<<<(unsigned)((n + 255) / 256), 256, 0, stream>>>(s, dH, dL, n);
  };
  splitLaunch(projw, projwH, projwL, NPROJW);
  splitLaunch(qkvw,  qkvwH,  qkvwL,  NQKVW);
  splitLaunch(rw,    rwwH,   rwwL,   NRWW);
  splitLaunch(ow,    owwH,   owwL,   NOWW);
  splitLaunch(ff1w,  ff1wH,  ff1wL,  NFF1);
  splitLaunch(ff2w,  ff2wH,  ff2wL,  NFF2);

  // ---- embedding + input projection + positions ----
  k_embed<<<(QL * BS * 768 + 255) / 256, 256, 0, stream>>>(pitch, dur, chord, pew, dew, embH, embL);
  k_gemm16<false, false><<<dim3(DMODEL / 32, QL * BS / 32), 32, 0, stream>>>(
      embH, embL, projwH, projwL, projb, h, nullptr, nullptr, QL * BS, DMODEL, 768);
  k_pos<<<(KL * 256 + 255) / 256, 256, 0, stream>>>(rH, rL);

  for (int l = 0; l < NLAYER; ++l) {
    size_t oq = (size_t)l * QKV3 * DMODEL;
    size_t os = (size_t)l * DMODEL * DMODEL;
    size_t of = (size_t)l * DIFF * DMODEL;
    k_cat<<<(KL * BS * DMODEL + 255) / 256, 256, 0, stream>>>(
        mems + (size_t)l * ML * BS * DMODEL, h, catH, catL);
    // w_heads = cat @ qkv_w[l]^T  -> planes
    k_gemm16<false, true><<<dim3(QKV3 / 32, KL * BS / 32), 32, 0, stream>>>(
        catH, catL, qkvwH + oq, qkvwL + oq, nullptr, nullptr, whH, whL, KL * BS, QKV3, DMODEL);
    // r_head_k = r @ r_w[l]^T -> planes
    k_gemm16<false, true><<<dim3(DMODEL / 32, KL / 32), 32, 0, stream>>>(
        rH, rL, rwwH + os, rwwL + os, nullptr, nullptr, rkH, rkL, KL, DMODEL, DMODEL);
    // fused relative attention
    k_attn<<<dim3(QL / 16, BS, NHEAD), 32, 0, stream>>>(whH, whL, rkH, rkL, rwb, rrb, avH, avL);
    // attn_out = attn_vec @ o_w[l]^T -> f32 tmp
    k_gemm16<false, false><<<dim3(DMODEL / 32, QL * BS / 32), 32, 0, stream>>>(
        avH, avL, owwH + os, owwL + os, nullptr, tmp, nullptr, nullptr, QL * BS, DMODEL, DMODEL);
    k_addln<<<QL * BS, 32, 0, stream>>>(h, tmp, ln1g + l * DMODEL, ln1b + l * DMODEL, hbH, hbL);
    // FFN
    k_gemm16<true, true><<<dim3(DIFF / 32, QL * BS / 32), 32, 0, stream>>>(
        hbH, hbL, ff1wH + of, ff1wL + of, ff1b + l * DIFF, nullptr, ffmH, ffmL, QL * BS, DIFF, DMODEL);
    k_gemm16<false, false><<<dim3(DMODEL / 32, QL * BS / 32), 32, 0, stream>>>(
        ffmH, ffmL, ff2wH + of, ff2wL + of, ff2b + l * DMODEL, tmp, nullptr, nullptr, QL * BS, DMODEL, DIFF);
    k_addln<<<QL * BS, 32, 0, stream>>>(h, tmp, ln2g + l * DMODEL, ln2b + l * DMODEL, hbH, hbL);
  }
}
